// DeeperGCN_70325794504883
// MI455X (gfx1250) — matrix operand order repaired and checked by reading, not compiled, not run on hardware
//
#include <hip/hip_runtime.h>
#include <hip/hip_bf16.h>

// Problem constants (match reference)
#define NN 50000
#define EE 600000
#define DD 128
#define D2 256
#define LL 7
#define BB 512
#define TTK 10
#define MSG_EPS 1e-7f
#define BN_EPS 1e-5f

typedef __attribute__((ext_vector_type(16))) __bf16 v16bf;
typedef __attribute__((ext_vector_type(8)))  float  v8f;

typedef union { unsigned u[8]; v16bf v; } Frag;

// ---------- helpers ----------
__device__ __forceinline__ unsigned short f2bf(float f) {
    unsigned u = __float_as_uint(f);
    unsigned r = u + 0x7FFFu + ((u >> 16) & 1u);   // round-to-nearest-even
    return (unsigned short)(r >> 16);
}

// ---------- encoders ----------
__global__ void atom_encoder_kernel(const int* __restrict__ x, const int* __restrict__ offs,
                                    const float* __restrict__ table, float* __restrict__ h) {
    int idx = blockIdx.x * blockDim.x + threadIdx.x;
    if (idx >= NN * DD) return;
    int n = idx >> 7, d = idx & 127;
    float s = 0.f;
#pragma unroll
    for (int i = 0; i < 9; ++i) {
        int row = x[n * 9 + i] + offs[i];
        s += table[(size_t)row * DD + d];
    }
    h[idx] = s;
}

__global__ void bond_encoder_kernel(const int* __restrict__ ea_idx, const int* __restrict__ offs,
                                    const float* __restrict__ table, float* __restrict__ ea) {
    size_t idx = (size_t)blockIdx.x * blockDim.x + threadIdx.x;
    if (idx >= (size_t)EE * DD) return;
    size_t e = idx >> 7; int d = (int)(idx & 127);
    float s = 0.f;
#pragma unroll
    for (int i = 0; i < 3; ++i) {
        int row = ea_idx[e * 3 + i] + offs[i];
        s += table[(size_t)row * DD + d];
    }
    ea[idx] = s;
}

// ---------- batchnorm (training-mode, biased var) ----------
// sums[0..cols) = sum, sums[cols..2*cols) = sum of squares. 64 rows per block.
__global__ void bn_stats_kernel(const float* __restrict__ X, int rows, int cols,
                                float* __restrict__ sums) {
    int col = threadIdx.x;              // blockDim.x == cols
    int r0 = blockIdx.x * 64;
    float s = 0.f, ss = 0.f;
    for (int i = 0; i < 64; ++i) {
        int r = r0 + i;
        if (r < rows) {
            float v = X[(size_t)r * cols + col];
            s += v; ss += v * v;
        }
    }
    atomicAdd(&sums[col], s);
    atomicAdd(&sums[cols + col], ss);
}

__global__ void bn_relu_f32_kernel(const float* __restrict__ X, const float* __restrict__ sums,
                                   const float* __restrict__ gamma, const float* __restrict__ beta,
                                   float* __restrict__ Y, int rows, int cols) {
    size_t idx = (size_t)blockIdx.x * blockDim.x + threadIdx.x;
    if (idx >= (size_t)rows * cols) return;
    int col = (int)(idx % cols);
    float inv_n = 1.f / (float)rows;
    float mu = sums[col] * inv_n;
    float var = sums[cols + col] * inv_n - mu * mu;
    float v = (X[idx] - mu) * rsqrtf(var + BN_EPS) * gamma[col] + beta[col];
    Y[idx] = v > 0.f ? v : 0.f;
}

__global__ void bn_relu_bf16_kernel(const float* __restrict__ X, const float* __restrict__ sums,
                                    const float* __restrict__ gamma, const float* __restrict__ beta,
                                    unsigned short* __restrict__ Y, int rows, int cols) {
    size_t idx = (size_t)blockIdx.x * blockDim.x + threadIdx.x;
    if (idx >= (size_t)rows * cols) return;
    int col = (int)(idx % cols);
    float inv_n = 1.f / (float)rows;
    float mu = sums[col] * inv_n;
    float var = sums[cols + col] * inv_n - mu * mu;
    float v = (X[idx] - mu) * rsqrtf(var + BN_EPS) * gamma[col] + beta[col];
    Y[idx] = f2bf(v > 0.f ? v : 0.f);
}

// ---------- GENConv softmax aggregation ----------
// Pass 1: per-(dst,channel) max of logits. Logits > 0 so float bits compare as unsigned.
__global__ void msg_max_kernel(const float* __restrict__ r, const float* __restrict__ ea,
                               const int* __restrict__ ei, const float* __restrict__ t_all,
                               int layer, unsigned* __restrict__ maxb) {
    int e = blockIdx.x, d = threadIdx.x;
    int s  = ei[e];
    int dn = ei[EE + e];
    float t = t_all[layer];
    float m = r[(size_t)s * DD + d] + ea[(size_t)e * DD + d];
    m = (m > 0.f ? m : 0.f) + MSG_EPS;
    float logit = m * t;
    atomicMax(&maxb[(size_t)dn * DD + d], __float_as_uint(logit));
}

// Pass 2: recompute message, accumulate exp-weights and weighted messages.
__global__ void msg_sum_kernel(const float* __restrict__ r, const float* __restrict__ ea,
                               const int* __restrict__ ei, const float* __restrict__ t_all,
                               int layer, const unsigned* __restrict__ maxb,
                               float* __restrict__ den, float* __restrict__ num) {
    int e = blockIdx.x, d = threadIdx.x;
    int s  = ei[e];
    int dn = ei[EE + e];
    float t = t_all[layer];
    float m = r[(size_t)s * DD + d] + ea[(size_t)e * DD + d];
    m = (m > 0.f ? m : 0.f) + MSG_EPS;
    float logit = m * t;
    float mx = __uint_as_float(maxb[(size_t)dn * DD + d]);
    float w = __expf(logit - mx);
    atomicAdd(&den[(size_t)dn * DD + d], w);
    atomicAdd(&num[(size_t)dn * DD + d], m * w);
}

// u = conv_input + aggr, emitted as bf16 (GEMM1 A operand)
__global__ void aggr_finalize_kernel(const float* __restrict__ r, const float* __restrict__ num,
                                     const float* __restrict__ den, unsigned short* __restrict__ u_bf) {
    size_t idx = (size_t)blockIdx.x * blockDim.x + threadIdx.x;
    if (idx >= (size_t)NN * DD) return;
    float aggr = num[idx] / (den[idx] + 1e-16f);
    u_bf[idx] = f2bf(r[idx] + aggr);
}

// W [K,Nout] f32 -> WT [Nout,K] bf16 (so the B-fragment K-pairs are contiguous dwords)
__global__ void weight_transpose_bf16_kernel(const float* __restrict__ W, unsigned short* __restrict__ WT,
                                             int K, int Nout) {
    int idx = blockIdx.x * blockDim.x + threadIdx.x;
    if (idx >= K * Nout) return;
    int n = idx / K, k = idx % K;
    WT[idx] = f2bf(W[(size_t)k * Nout + n]);
}

// ---------- WMMA GEMM: C[M,Nout] (+)= A[M,K]_bf16 @ B[K,Nout]_bf16 + bias ----------
// One wave32 owns a 16x(16*NB) output block: the A fragment is loaded once per K-step
// and reused across NB WMMAs (4x less A traffic than one-tile-per-wave).
template <int NB>
__global__ void gemm_bf16_wmma_kernel(const unsigned short* __restrict__ A,
                                      const unsigned short* __restrict__ BT, // [Nout,K]
                                      const float* __restrict__ bias,
                                      float* __restrict__ C,
                                      int M, int K, int Nout, int accumulate) {
    int wavesPerBlock = blockDim.x >> 5;
    int wave = blockIdx.x * wavesPerBlock + ((int)threadIdx.x >> 5);
    int groupsN = (Nout >> 4) / NB;
    int total = (M >> 4) * groupsN;
    if (wave >= total) return;          // whole-wave uniform: EXEC stays all-ones for WMMA
    int tm = wave / groupsN, tg = wave % groupsN;
    int lane   = threadIdx.x & 31;
    int laneHi = lane >> 4;
    int lane16 = lane & 15;

    const unsigned short* Arow = A + (size_t)((tm << 4) + lane16) * K;   // A row-major
    const unsigned short* Bcol[NB];
#pragma unroll
    for (int j = 0; j < NB; ++j)
        Bcol[j] = BT + (size_t)(((tg * NB + j) << 4) + lane16) * K;      // B col (= WT row)

    // ISA 16-bit A 16x32 layout: lanes0-15 VGPR0-3:K0-7, VGPR4-7:K16-23; lanes16-31: +8
    // 16-bit B 32x16 layout (same k order as A): lanes0-15 VGPR0-3:K0-7, VGPR4-7:K16-23; lanes16-31: +8
    int kA[8], kB[8];
#pragma unroll
    for (int v = 0; v < 8; ++v) {
        kA[v] = ((v < 4) ? (2 * v) : (2 * v + 8)) + (laneHi ? 8 : 0);
        kB[v] = ((v < 4) ? (2 * v) : (2 * v + 8)) + (laneHi ? 8 : 0);
    }

    v8f acc[NB];
#pragma unroll
    for (int j = 0; j < NB; ++j) acc[j] = (v8f){};

    for (int k0 = 0; k0 < K; k0 += 32) {
        Frag fa;
#pragma unroll
        for (int v = 0; v < 8; ++v)
            fa.u[v] = *(const unsigned*)(Arow + k0 + kA[v]);
        Frag fb[NB];
#pragma unroll
        for (int j = 0; j < NB; ++j)
#pragma unroll
            for (int v = 0; v < 8; ++v)
                fb[j].u[v] = *(const unsigned*)(Bcol[j] + k0 + kB[v]);
#pragma unroll
        for (int j = 0; j < NB; ++j)
            acc[j] = __builtin_amdgcn_wmma_f32_16x16x32_bf16(
                /*neg_a=*/false, fa.v, /*neg_b=*/false, fb[j].v,
                /*c_mod=*/(short)0, acc[j], /*reuse_a=*/false, /*reuse_b=*/false);
    }

    // C/D layout: VGPR j -> rows j (lanes0-15) / j+8 (lanes16-31), col = lane16
#pragma unroll
    for (int j = 0; j < NB; ++j) {
        int col = ((tg * NB + j) << 4) + lane16;
        float b = bias[col];
#pragma unroll
        for (int jj = 0; jj < 8; ++jj) {
            int row = (tm << 4) + jj + (laneHi << 3);
            size_t off = (size_t)row * Nout + col;
            float val = acc[j][jj] + b;
            if (accumulate) C[off] += val; else C[off] = val;
        }
    }
}

// ---------- readout ----------
__global__ void final_bn_pool_kernel(const float* __restrict__ h, const float* __restrict__ sums,
                                     const float* __restrict__ gamma, const float* __restrict__ beta,
                                     const int* __restrict__ batch,
                                     float* __restrict__ pool, float* __restrict__ cnt) {
    size_t idx = (size_t)blockIdx.x * blockDim.x + threadIdx.x;
    if (idx >= (size_t)NN * DD) return;
    int n = (int)(idx >> 7), d = (int)(idx & 127);
    float inv_n = 1.f / (float)NN;
    float mu = sums[d] * inv_n;
    float var = sums[DD + d] * inv_n - mu * mu;
    float v = (h[idx] - mu) * rsqrtf(var + BN_EPS) * gamma[d] + beta[d];
    int g = batch[n];
    atomicAdd(&pool[(size_t)g * DD + d], v);
    if (d == 0) atomicAdd(&cnt[g], 1.f);
}

__global__ void predict_kernel(const float* __restrict__ pool, const float* __restrict__ cnt,
                               const float* __restrict__ W, const float* __restrict__ b,
                               float* __restrict__ out) {
    int idx = blockIdx.x * blockDim.x + threadIdx.x;
    if (idx >= BB * TTK) return;
    int g = idx / TTK, t = idx % TTK;
    float c = cnt[g]; c = c > 1.f ? c : 1.f;
    float inv_c = 1.f / c;
    float s = b[t];
#pragma unroll 8
    for (int d = 0; d < DD; ++d)
        s += pool[(size_t)g * DD + d] * inv_c * W[d * TTK + t];
    out[idx] = s;
}

// ---------- host orchestration ----------
extern "C" void kernel_launch(void* const* d_in, const int* in_sizes, int n_in,
                              void* d_out, int out_size, void* d_ws, size_t ws_size,
                              hipStream_t stream) {
    (void)in_sizes; (void)n_in; (void)out_size; (void)ws_size;
    // inputs per setup_inputs() order
    const int*   x            = (const int*)d_in[0];
    const int*   edge_index   = (const int*)d_in[1];
    const int*   edge_attr    = (const int*)d_in[2];
    const int*   batch        = (const int*)d_in[3];
    const int*   atom_offs    = (const int*)d_in[4];
    const int*   bond_offs    = (const int*)d_in[5];
    const float* atom_table   = (const float*)d_in[6];
    const float* bond_table   = (const float*)d_in[7];
    const float* norm_gamma   = (const float*)d_in[8];
    const float* norm_beta    = (const float*)d_in[9];
    const float* t_all        = (const float*)d_in[10];
    const float* mlp_w1       = (const float*)d_in[11];
    const float* mlp_b1       = (const float*)d_in[12];
    const float* mlp_bng      = (const float*)d_in[13];
    const float* mlp_bnb      = (const float*)d_in[14];
    const float* mlp_w2       = (const float*)d_in[15];
    const float* mlp_b2       = (const float*)d_in[16];
    const float* pred_w       = (const float*)d_in[17];
    const float* pred_b       = (const float*)d_in[18];
    float* out = (float*)d_out;

    // workspace layout (~525 MB)
    char* base = (char*)d_ws;
    size_t o = 0;
    auto take = [&](size_t bytes) -> void* {
        void* p = base + o;
        o += (bytes + 255) & ~(size_t)255;
        return p;
    };
    float*          ea    = (float*)take((size_t)EE * DD * 4);
    float*          h     = (float*)take((size_t)NN * DD * 4);
    float*          r     = (float*)take((size_t)NN * DD * 4);
    unsigned*       maxb  = (unsigned*)take((size_t)NN * DD * 4);
    float*          den   = (float*)take((size_t)NN * DD * 4);
    float*          num   = (float*)take((size_t)NN * DD * 4);
    float*          z     = (float*)take((size_t)NN * D2 * 4);
    unsigned short* u_bf  = (unsigned short*)take((size_t)NN * DD * 2);
    unsigned short* zr_bf = (unsigned short*)take((size_t)NN * D2 * 2);
    unsigned short* w1_bf = (unsigned short*)take((size_t)DD * D2 * 2);
    unsigned short* w2_bf = (unsigned short*)take((size_t)D2 * DD * 2);
    float*          stats = (float*)take(2 * D2 * 4);
    float*          pool  = (float*)take((size_t)BB * DD * 4);
    float*          cnt   = (float*)take((size_t)BB * 4);

    const int ND  = NN * DD;            // 6.4M
    const int ND2 = NN * D2;            // 12.8M
    const int bND  = (ND  + 255) / 256;
    const int bND2 = (ND2 + 255) / 256;
    const int statBlocks = (NN + 63) / 64;

    // encoders (run once; ea reused by all 7 layers)
    atom_encoder_kernel<<<bND, 256, 0, stream>>>(x, atom_offs, atom_table, h);
    {
        size_t tot = (size_t)EE * DD;
        bond_encoder_kernel<<<(unsigned)((tot + 255) / 256), 256, 0, stream>>>(edge_attr, bond_offs, bond_table, ea);
    }

    for (int layer = 0; layer < LL; ++layer) {
        const float* g1  = mlp_bng + layer * D2;
        const float* bt1 = mlp_bnb + layer * D2;
        const float* b1  = mlp_b1  + layer * D2;
        const float* b2  = mlp_b2  + layer * DD;
        const float* W1  = mlp_w1  + (size_t)layer * DD * D2;
        const float* W2  = mlp_w2  + (size_t)layer * D2 * DD;

        const float* conv_in = h;
        if (layer > 0) {
            // res+ pre-norm: r = relu(BN(h))
            hipMemsetAsync(stats, 0, 2 * DD * 4, stream);
            bn_stats_kernel<<<statBlocks, DD, 0, stream>>>(h, NN, DD, stats);
            bn_relu_f32_kernel<<<bND, 256, 0, stream>>>(h, stats,
                norm_gamma + layer * DD, norm_beta + layer * DD, r, NN, DD);
            conv_in = r;
        }

        // softmax aggregation (2 edge passes + finalize)
        hipMemsetAsync(maxb, 0, (size_t)ND * 4, stream);
        hipMemsetAsync(den,  0, (size_t)ND * 4, stream);
        hipMemsetAsync(num,  0, (size_t)ND * 4, stream);
        msg_max_kernel<<<EE, DD, 0, stream>>>(conv_in, ea, edge_index, t_all, layer, maxb);
        msg_sum_kernel<<<EE, DD, 0, stream>>>(conv_in, ea, edge_index, t_all, layer, maxb, den, num);
        aggr_finalize_kernel<<<bND, 256, 0, stream>>>(conv_in, num, den, u_bf);

        // per-layer weight prep (tiny)
        weight_transpose_bf16_kernel<<<(DD * D2 + 255) / 256, 256, 0, stream>>>(W1, w1_bf, DD, D2);
        weight_transpose_bf16_kernel<<<(D2 * DD + 255) / 256, 256, 0, stream>>>(W2, w2_bf, D2, DD);

        // GEMM1: z[N,256] = u @ W1 + b1   (3125 row tiles x 4 col groups, 8 waves/block)
        {
            int waves = (NN / 16) * ((D2 / 16) / 4);
            gemm_bf16_wmma_kernel<4><<<(waves + 7) / 8, 256, 0, stream>>>(u_bf, w1_bf, b1, z, NN, DD, D2, 0);
        }

        // MLP BN + ReLU (batch stats over N, 256 channels) -> bf16
        hipMemsetAsync(stats, 0, 2 * D2 * 4, stream);
        bn_stats_kernel<<<statBlocks, D2, 0, stream>>>(z, NN, D2, stats);
        bn_relu_bf16_kernel<<<bND2, 256, 0, stream>>>(z, stats, g1, bt1, zr_bf, NN, D2);

        // GEMM2: h (=|+=) relu(z) @ W2 + b2  (fused residual)
        {
            int waves = (NN / 16) * ((DD / 16) / 4);
            gemm_bf16_wmma_kernel<4><<<(waves + 7) / 8, 256, 0, stream>>>(zr_bf, w2_bf, b2, h, NN, D2, DD,
                                                                          layer > 0 ? 1 : 0);
        }
    }

    // final BN (layers[0].norm) + global mean pool + prediction head
    hipMemsetAsync(stats, 0, 2 * DD * 4, stream);
    bn_stats_kernel<<<statBlocks, DD, 0, stream>>>(h, NN, DD, stats);
    hipMemsetAsync(pool, 0, (size_t)BB * DD * 4, stream);
    hipMemsetAsync(cnt,  0, (size_t)BB * 4, stream);
    final_bn_pool_kernel<<<bND, 256, 0, stream>>>(h, stats, norm_gamma, norm_beta, batch, pool, cnt);
    predict_kernel<<<(BB * TTK + 255) / 256, 256, 0, stream>>>(pool, cnt, pred_w, pred_b, out);
}
